// CausalSelfAttention_18030272709193
// MI455X (gfx1250) — matrix operand order repaired and checked by reading, not compiled, not run on hardware
//
#include <hip/hip_runtime.h>
#include <stdint.h>

// ---------- WMMA fragment types (gfx1250, wave32) ----------
typedef __bf16 v16bf __attribute__((ext_vector_type(16)));
typedef float  v8f   __attribute__((ext_vector_type(8)));

union FragU { uint32_t u[8]; v16bf v; };
union U4U   { uint4 q; uint16_t s[8]; };

__device__ __forceinline__ v8f zero8() {
  v8f z = {0.f,0.f,0.f,0.f,0.f,0.f,0.f,0.f};
  return z;
}

__device__ __forceinline__ uint16_t f2bf(float f) {
  union { float f; uint32_t u; } c; c.f = f;
  uint32_t u = c.u + 0x7FFFu + ((c.u >> 16) & 1u);   // RNE
  return (uint16_t)(u >> 16);
}

__device__ __forceinline__ v8f wmma_bf16(v16bf a, v16bf b, v8f c) {
  // (neg_a, A, neg_b, B, c_mod, C, reuse_a, reuse_b)
  return __builtin_amdgcn_wmma_f32_16x16x32_bf16(false, a, false, b, (short)0, c,
                                                 false, false);
}

// A-matrix 16x32 bf16 fragment from row-major LDS tile.
// ISA layout: lanes 0-15 row M=lane, K elems {0..7,16..23}; lanes 16-31 K+=8.
__device__ __forceinline__ v16bf lds_frag_a(const uint16_t* s, int stride,
                                            int rowbase, int kbase, int lane) {
  const int row = rowbase + (lane & 15);
  const uint16_t* p = s + row * stride + kbase + ((lane >> 4) << 3);
  FragU f;
#pragma unroll
  for (int j = 0; j < 4; ++j) f.u[j]     = *(const uint32_t*)(p + 2 * j);
#pragma unroll
  for (int j = 0; j < 4; ++j) f.u[4 + j] = *(const uint32_t*)(p + 16 + 2 * j);
  return f.v;
}

// B-matrix 32x16 bf16 fragment from LDS stored [N][K] (K contiguous).
// ISA layout: lane holds col N=lane&15; lanes 0-15 K elems {0..7,16..23}; lanes 16-31 K+=8.
__device__ __forceinline__ v16bf lds_frag_b(const uint16_t* s, int stride,
                                            int colbase, int kbase, int lane) {
  const int col = colbase + (lane & 15);
  const uint16_t* p = s + col * stride + kbase + ((lane >> 4) << 3);
  FragU f;
#pragma unroll
  for (int j = 0; j < 4; ++j) f.u[j]     = *(const uint32_t*)(p + 2 * j);
#pragma unroll
  for (int j = 0; j < 4; ++j) f.u[4 + j] = *(const uint32_t*)(p + 16 + 2 * j);
  return f.v;
}

// ---------- f32 -> bf16 convert ----------
__global__ void cvt_bf16(const float* __restrict__ in, uint16_t* __restrict__ out,
                         long n) {
  long i = ((long)blockIdx.x * 256 + threadIdx.x) * 4;
  if (i >= n) return;
  float4 v = *(const float4*)(in + i);
  union { uint16_t h[4]; uint32_t u[2]; } pk;
  pk.h[0] = f2bf(v.x); pk.h[1] = f2bf(v.y); pk.h[2] = f2bf(v.z); pk.h[3] = f2bf(v.w);
  *(uint32_t*)(out + i)     = pk.u[0];
  *(uint32_t*)(out + i + 2) = pk.u[1];
}

// ---------- bf16 WMMA GEMM: C[M,N](f32) = A[M,K] * B[K,N], row-major ----------
// Block 128x128, 8 waves (2x4), wave tile 64x32, K-step 64 (2 WMMA k-slices).
__global__ __launch_bounds__(256)
void gemm_bf16(const uint16_t* __restrict__ A, const uint16_t* __restrict__ B,
               float* __restrict__ C, int M, int N, int K) {
  __shared__ uint16_t As[128][72];   // 144B rows: 16B aligned, 36-dword bank stride
  __shared__ uint16_t Bt[128][72];   // B staged transposed: [n][k]
  const int tid = threadIdx.x, w = tid >> 5, lane = tid & 31;
  const int bm = blockIdx.y, bn = blockIdx.x;
  const int waveM = w >> 2, waveN = w & 3;
  v8f acc[4][2];
#pragma unroll
  for (int i = 0; i < 4; ++i)
#pragma unroll
    for (int j = 0; j < 2; ++j) acc[i][j] = zero8();

  for (int kt = 0; kt < K; kt += 64) {
    __syncthreads();
#pragma unroll
    for (int i = 0; i < 4; ++i) {                 // stage A 128x64
      int idx = tid + i * 256;
      int r = idx >> 3, c = (idx & 7) * 8;
      *(uint4*)(&As[r][c]) = *(const uint4*)(A + (size_t)(bm * 128 + r) * K + kt + c);
    }
#pragma unroll
    for (int i = 0; i < 4; ++i) {                 // stage B 64x128, transposed
      int idx = tid + i * 256;
      int kr = idx >> 4, c0 = (idx & 15) * 8;
      U4U d; d.q = *(const uint4*)(B + (size_t)(kt + kr) * N + bn * 128 + c0);
#pragma unroll
      for (int j = 0; j < 8; ++j) Bt[c0 + j][kr] = d.s[j];
    }
    if (kt + 64 < K) {                            // global_prefetch_b8 next tiles
      __builtin_prefetch(A + (size_t)(bm * 128 + (tid >> 1)) * K + kt + 64, 0, 0);
      __builtin_prefetch(B + (size_t)(kt + 64 + (tid >> 2)) * N + bn * 128, 0, 0);
    }
    __syncthreads();
#pragma unroll
    for (int kb = 0; kb < 64; kb += 32) {
      v16bf af[4], bfm[2];
#pragma unroll
      for (int mf = 0; mf < 4; ++mf)
        af[mf] = lds_frag_a(&As[0][0], 72, waveM * 64 + mf * 16, kb, lane);
#pragma unroll
      for (int nf = 0; nf < 2; ++nf)
        bfm[nf] = lds_frag_b(&Bt[0][0], 72, waveN * 32 + nf * 16, kb, lane);
#pragma unroll
      for (int mf = 0; mf < 4; ++mf)
#pragma unroll
        for (int nf = 0; nf < 2; ++nf)
          acc[mf][nf] = wmma_bf16(af[mf], bfm[nf], acc[mf][nf]);
    }
  }
  const int khalf = lane >> 4, nlo = lane & 15;
#pragma unroll
  for (int mf = 0; mf < 4; ++mf)
#pragma unroll
    for (int nf = 0; nf < 2; ++nf)
#pragma unroll
      for (int m = 0; m < 8; ++m) {
        int row = bm * 128 + waveM * 64 + mf * 16 + m + khalf * 8;
        int col = bn * 128 + waveN * 32 + nf * 16 + nlo;
        C[(size_t)row * N + col] = acc[mf][nf][m];
      }
}

// ---------- rmsnorm over 128-element head segments (one wave per segment) ----
// outn = bf16(rmsnorm(x)); optional outp = bf16(x)
__global__ __launch_bounds__(256)
void rms128(const float* __restrict__ in, uint16_t* __restrict__ outn,
            uint16_t* __restrict__ outp, int nseg) {
  const int w = threadIdx.x >> 5, lane = threadIdx.x & 31;
  const int seg = blockIdx.x * 8 + w;
  if (seg >= nseg) return;
  const float* p = in + (size_t)seg * 128 + lane * 4;
  float4 v = *(const float4*)p;
  float s = v.x * v.x + v.y * v.y + v.z * v.z + v.w * v.w;
  s += __shfl_xor(s, 1, 32);  s += __shfl_xor(s, 2, 32);
  s += __shfl_xor(s, 4, 32);  s += __shfl_xor(s, 8, 32);
  s += __shfl_xor(s, 16, 32);
  const float scale = rsqrtf(s * (1.f / 128.f) + 1.1920929e-07f);
  union { uint16_t h[4]; uint32_t u[2]; } pk;
  uint16_t* q = outn + (size_t)seg * 128 + lane * 4;
  pk.h[0] = f2bf(v.x * scale); pk.h[1] = f2bf(v.y * scale);
  pk.h[2] = f2bf(v.z * scale); pk.h[3] = f2bf(v.w * scale);
  *(uint32_t*)(q) = pk.u[0];  *(uint32_t*)(q + 2) = pk.u[1];
  if (outp) {
    uint16_t* q2 = outp + (size_t)seg * 128 + lane * 4;
    pk.h[0] = f2bf(v.x); pk.h[1] = f2bf(v.y); pk.h[2] = f2bf(v.z); pk.h[3] = f2bf(v.w);
    *(uint32_t*)(q2) = pk.u[0];  *(uint32_t*)(q2 + 2) = pk.u[1];
  }
}

// ---------- flash attention (no mask; reference mask is all-False) ----------
// grid (T/64, HQ, B); 4 waves; Q tile 64x128, K/V tiles 32x128.
__global__ __launch_bounds__(128)
void attn_fwd(const uint16_t* __restrict__ Qn, const uint16_t* __restrict__ Kn,
              const uint16_t* __restrict__ Vb, uint16_t* __restrict__ Y) {
  __shared__ uint16_t Qs[64][136];     // 272B rows (16B aligned)
  __shared__ uint16_t Ks[32][136];     // K tile row-major = B-layout for Q*K^T
  __shared__ uint16_t Vt[128][40];     // V tile transposed [d][s] for P*V
  __shared__ uint16_t Ps[4][16][40];   // per-wave P relayout C->A
  const int tid = threadIdx.x, w = tid >> 5, lane = tid & 31;
  const int khalf = lane >> 4, nlo = lane & 15;
  const int qt = blockIdx.x, h = blockIdx.y, b = blockIdx.z;
  const int hk = h >> 2;               // GQA group of 4
  const uint16_t* Qbase = Qn + (size_t)(b * 2048 + qt * 64) * 2048 + h * 128;
  const uint16_t* Kbase = Kn + (size_t)(b * 2048) * 512 + hk * 128;
  const uint16_t* Vbase = Vb + (size_t)(b * 2048) * 512 + hk * 128;

#pragma unroll
  for (int i = 0; i < 8; ++i) {        // stage Q tile once
    int idx = tid + i * 128;
    int r = idx >> 4, c = (idx & 15) * 8;
    *(uint4*)(&Qs[r][c]) = *(const uint4*)(Qbase + (size_t)r * 2048 + c);
  }
  __syncthreads();
  v16bf qa[4];
#pragma unroll
  for (int kb = 0; kb < 4; ++kb)       // Q A-frags live in regs for whole loop
    qa[kb] = lds_frag_a(&Qs[0][0], 136, w * 16, kb * 32, lane);

  float m_run[8], l_run[8];
  v8f o[8];
#pragma unroll
  for (int m = 0; m < 8; ++m) { m_run[m] = -1e30f; l_run[m] = 0.f; }
#pragma unroll
  for (int f = 0; f < 8; ++f) o[f] = zero8();

  for (int s0 = 0; s0 < 2048; s0 += 32) {
    __syncthreads();
#pragma unroll
    for (int i = 0; i < 4; ++i) {      // stage K (as-is) and V (transposed)
      int idx = tid + i * 128;
      int r = idx >> 4, c = (idx & 15) * 8;
      *(uint4*)(&Ks[r][c]) = *(const uint4*)(Kbase + (size_t)(s0 + r) * 512 + c);
      U4U d; d.q = *(const uint4*)(Vbase + (size_t)(s0 + r) * 512 + c);
#pragma unroll
      for (int j = 0; j < 8; ++j) Vt[c + j][r] = d.s[j];
    }
    __syncthreads();

    v8f sf0 = zero8(), sf1 = zero8();  // S = Q * K^T   (16x32 per wave)
#pragma unroll
    for (int kb = 0; kb < 4; ++kb) {
      v16bf b0 = lds_frag_b(&Ks[0][0], 136, 0,  kb * 32, lane);
      v16bf b1 = lds_frag_b(&Ks[0][0], 136, 16, kb * 32, lane);
      sf0 = wmma_bf16(qa[kb], b0, sf0);
      sf1 = wmma_bf16(qa[kb], b1, sf1);
    }
    // online softmax (row = m + 8*khalf, reductions over each 16-lane half)
    float corr[8];
#pragma unroll
    for (int m = 0; m < 8; ++m) {
      float v = fmaxf(sf0[m], sf1[m]);
      v = fmaxf(v, __shfl_xor(v, 1, 32));
      v = fmaxf(v, __shfl_xor(v, 2, 32));
      v = fmaxf(v, __shfl_xor(v, 4, 32));
      v = fmaxf(v, __shfl_xor(v, 8, 32));
      float mn = fmaxf(m_run[m], v);
      corr[m] = __expf(m_run[m] - mn);
      m_run[m] = mn;
      float p0 = __expf(sf0[m] - mn);
      float p1 = __expf(sf1[m] - mn);
      sf0[m] = p0; sf1[m] = p1;
      float ps = p0 + p1;
      ps += __shfl_xor(ps, 1, 32);
      ps += __shfl_xor(ps, 2, 32);
      ps += __shfl_xor(ps, 4, 32);
      ps += __shfl_xor(ps, 8, 32);
      l_run[m] = l_run[m] * corr[m] + ps;
    }
#pragma unroll
    for (int f = 0; f < 8; ++f)
#pragma unroll
      for (int m = 0; m < 8; ++m) o[f][m] = o[f][m] * corr[m];
    // P: C-layout -> A-layout via per-wave LDS
#pragma unroll
    for (int m = 0; m < 8; ++m) {
      Ps[w][m + khalf * 8][nlo]      = f2bf(sf0[m]);
      Ps[w][m + khalf * 8][16 + nlo] = f2bf(sf1[m]);
    }
    __syncthreads();
    v16bf pa = lds_frag_a(&Ps[w][0][0], 40, 0, 0, lane);
#pragma unroll
    for (int f = 0; f < 8; ++f) {      // O += P * V
      v16bf bv = lds_frag_b(&Vt[0][0], 40, f * 16, 0, lane);
      o[f] = wmma_bf16(pa, bv, o[f]);
    }
  }
  uint16_t* Ybase = Y + (size_t)(b * 2048 + qt * 64 + w * 16) * 2048 + h * 128;
#pragma unroll
  for (int m = 0; m < 8; ++m) {
    const float inv = 1.0f / l_run[m];
    const int row = m + khalf * 8;
#pragma unroll
    for (int f = 0; f < 8; ++f)
      Ybase[(size_t)row * 2048 + f * 16 + nlo] = f2bf(o[f][m] * inv);
  }
}

// ---------- final rmsnorm over full rows of 2048, fp32 out ----------
__global__ __launch_bounds__(256)
void rms_row(const float* __restrict__ in, float* __restrict__ out) {
  __shared__ float red[8];
  const int tid = threadIdx.x, w = tid >> 5, lane = tid & 31;
  const float* p = in + (size_t)blockIdx.x * 2048 + tid * 8;
  float4 a = *(const float4*)p;
  float4 c = *(const float4*)(p + 4);
  float s = a.x * a.x + a.y * a.y + a.z * a.z + a.w * a.w +
            c.x * c.x + c.y * c.y + c.z * c.z + c.w * c.w;
  s += __shfl_xor(s, 1, 32);  s += __shfl_xor(s, 2, 32);
  s += __shfl_xor(s, 4, 32);  s += __shfl_xor(s, 8, 32);
  s += __shfl_xor(s, 16, 32);
  if (lane == 0) red[w] = s;
  __syncthreads();
  float tot = 0.f;
#pragma unroll
  for (int i = 0; i < 8; ++i) tot += red[i];
  const float scale = rsqrtf(tot * (1.f / 2048.f) + 1.1920929e-07f);
  float* q = out + (size_t)blockIdx.x * 2048 + tid * 8;
  q[0] = a.x * scale; q[1] = a.y * scale; q[2] = a.z * scale; q[3] = a.w * scale;
  q[4] = c.x * scale; q[5] = c.y * scale; q[6] = c.z * scale; q[7] = c.w * scale;
}

// ---------------------------------------------------------------------------
extern "C" void kernel_launch(void* const* d_in, const int* in_sizes, int n_in,
                              void* d_out, int out_size, void* d_ws, size_t ws_size,
                              hipStream_t stream) {
  (void)in_sizes; (void)n_in; (void)out_size; (void)ws_size;
  const float* x  = (const float*)d_in[0];
  const float* Wq = (const float*)d_in[1];
  // d_in[2] (Wk) is dead in the reference: K = rmsnorm(V). Skipped.
  const float* Wv = (const float*)d_in[3];
  const float* Wo = (const float*)d_in[4];
  float* out = (float*)d_out;

  char* ws = (char*)d_ws;                              // layout (bytes):
  uint16_t* xb  = (uint16_t*)(ws + 0);                 // 16 MiB (reused as y bf16)
  uint16_t* Wqb = (uint16_t*)(ws + 16777216);          //  8 MiB
  uint16_t* Wvb = (uint16_t*)(ws + 25165824);          //  2 MiB
  uint16_t* Wob = (uint16_t*)(ws + 27262976);          //  8 MiB
  float*    Qf  = (float*)   (ws + 35651584);          // 32 MiB (reused as z f32)
  float*    Vf  = (float*)   (ws + 69206016);          //  8 MiB
  uint16_t* Qnb = (uint16_t*)(ws + 77594624);          // 16 MiB
  uint16_t* Knb = (uint16_t*)(ws + 94371840);          //  4 MiB
  uint16_t* Vbb = (uint16_t*)(ws + 98566144);          //  4 MiB  (total ~98 MiB)
  uint16_t* yb  = xb;                                  // x dead after projections
  float*    zf  = Qf;                                  // Q raw dead after rms128

  cvt_bf16<<<8192, 256, 0, stream>>>(x,  xb,  8388608L);
  cvt_bf16<<<4096, 256, 0, stream>>>(Wq, Wqb, 4194304L);
  cvt_bf16<<<1024, 256, 0, stream>>>(Wv, Wvb, 1048576L);
  cvt_bf16<<<4096, 256, 0, stream>>>(Wo, Wob, 4194304L);

  gemm_bf16<<<dim3(16, 32), 256, 0, stream>>>(xb, Wqb, Qf, 4096, 2048, 2048);
  gemm_bf16<<<dim3(4, 32),  256, 0, stream>>>(xb, Wvb, Vf, 4096, 512, 2048);

  rms128<<<8192, 256, 0, stream>>>(Qf, Qnb, nullptr, 65536);   // Q = rmsnorm(Q)
  rms128<<<2048, 256, 0, stream>>>(Vf, Knb, Vbb, 16384);       // K = rmsnorm(V), V bf16

  attn_fwd<<<dim3(32, 16, 2), 128, 0, stream>>>(Qnb, Knb, Vbb, yb);

  gemm_bf16<<<dim3(16, 32), 256, 0, stream>>>(yb, Wob, zf, 4096, 2048, 2048);

  rms_row<<<4096, 256, 0, stream>>>(zf, out);
}
